// PerceiverAttention_37864431682129
// MI455X (gfx1250) — matrix operand order repaired and checked by reading, not compiled, not run on hardware
//
#include <hip/hip_runtime.h>

// ---------------------------------------------------------------------------
// PerceiverAttention for MI455X (gfx1250): bf16 WMMA pipeline, f32 accum.
// Weights pre-transposed to N-major so GEMM A/B fragments are contiguous
// global b128 loads (L2-resident, 192MB); attention uses 64-key LDS tiles,
// permlane16 VALU reductions for online softmax.
// ---------------------------------------------------------------------------

typedef __bf16 bf16_t;
typedef bf16_t v16bf __attribute__((ext_vector_type(16)));
typedef float  v8f   __attribute__((ext_vector_type(8)));

union Frag16 {
    v16bf  v;
    uint4  u[2];
    bf16_t e[16];
};

namespace cfg {
constexpr int kDim   = 1024;
constexpr int kHeads = 16;
constexpr int kDHead = 64;
constexpr int kB     = 4;
constexpr int kNx    = 4096;
constexpr int kNl    = 1024;
constexpr int kNs    = kNx + kNl;   // 5120
}

// ---------------------------------------------------------------------------
// 16-lane xor reductions. permlane16 keeps these in VALU (no DS traffic);
// our reduction groups are exactly the 16-lane rows permlane16 operates on.
// ---------------------------------------------------------------------------
#if __has_builtin(__builtin_amdgcn_permlane16)
__device__ __forceinline__ float lanexor16(float v, unsigned lo, unsigned hi) {
    unsigned u = __float_as_uint(v);
    unsigned r = __builtin_amdgcn_permlane16(u, u, lo, hi, false, false);
    return __uint_as_float(r);
}
__device__ __forceinline__ float red_max16(float v) {
    v = fmaxf(v, lanexor16(v, 0x67452301u, 0xEFCDAB89u));  // xor 1
    v = fmaxf(v, lanexor16(v, 0x54761032u, 0xDCFE98BAu));  // xor 2
    v = fmaxf(v, lanexor16(v, 0x32107654u, 0xBA98FEDCu));  // xor 4
    v = fmaxf(v, lanexor16(v, 0xFEDCBA98u, 0x76543210u));  // xor 8
    return v;
}
__device__ __forceinline__ float red_add16(float v) {
    v += lanexor16(v, 0x67452301u, 0xEFCDAB89u);
    v += lanexor16(v, 0x54761032u, 0xDCFE98BAu);
    v += lanexor16(v, 0x32107654u, 0xBA98FEDCu);
    v += lanexor16(v, 0xFEDCBA98u, 0x76543210u);
    return v;
}
#else
__device__ __forceinline__ float red_max16(float v) {
    #pragma unroll
    for (int off = 8; off >= 1; off >>= 1) v = fmaxf(v, __shfl_xor(v, off, 32));
    return v;
}
__device__ __forceinline__ float red_add16(float v) {
    #pragma unroll
    for (int off = 8; off >= 1; off >>= 1) v += __shfl_xor(v, off, 32);
    return v;
}
#endif

// ---------------------------------------------------------------------------
// Weight transpose + convert: Wt[n][k] = (bf16)W[k][n]
// ---------------------------------------------------------------------------
__global__ __launch_bounds__(256)
void transpose_f32_bf16_kernel(const float* __restrict__ src, bf16_t* __restrict__ dst,
                               int K, int N) {
    __shared__ float t[32][33];
    const int k0 = blockIdx.x * 32, n0 = blockIdx.y * 32;
    const int tx = threadIdx.x & 31, ty = threadIdx.x >> 5;   // 32 x 8
    #pragma unroll
    for (int j = 0; j < 32; j += 8)
        t[ty + j][tx] = src[(size_t)(k0 + ty + j) * N + n0 + tx];
    __syncthreads();
    #pragma unroll
    for (int j = 0; j < 32; j += 8)
        dst[(size_t)(n0 + ty + j) * K + k0 + tx] = (bf16_t)t[tx][ty + j];
}

// ---------------------------------------------------------------------------
// LayerNorm over rows of kDim, emit bf16 to one or two destinations.
// ---------------------------------------------------------------------------
__global__ __launch_bounds__(256)
void ln_to_bf16_kernel(const float* __restrict__ src,
                       const float* __restrict__ gamma,
                       const float* __restrict__ beta,
                       bf16_t* __restrict__ dstA, int batchStrideA, int rowOffA,
                       bf16_t* __restrict__ dstB, int batchStrideB,
                       int rowsPerBatch) {
    using namespace cfg;
    const int row   = blockIdx.x;
    const int batch = row / rowsPerBatch;
    const int r     = row - batch * rowsPerBatch;
    const int tid   = threadIdx.x;

    const float* p = src + (size_t)row * kDim;
    float4 xv = *(const float4*)(p + tid * 4);
    float s  = xv.x + xv.y + xv.z + xv.w;
    float sq = xv.x * xv.x + xv.y * xv.y + xv.z * xv.z + xv.w * xv.w;

    #pragma unroll
    for (int off = 16; off >= 1; off >>= 1) {
        s  += __shfl_xor(s, off, 32);
        sq += __shfl_xor(sq, off, 32);
    }
    __shared__ float red[16];
    const int wid = tid >> 5, lane = tid & 31;
    if (lane == 0) { red[wid] = s; red[8 + wid] = sq; }
    __syncthreads();
    if (tid == 0) {
        float ts = 0.f, tq = 0.f;
        #pragma unroll
        for (int i = 0; i < 8; ++i) { ts += red[i]; tq += red[8 + i]; }
        red[0] = ts;
        red[1] = tq;
    }
    __syncthreads();
    const float mean = red[0] * (1.0f / kDim);
    const float var  = red[1] * (1.0f / kDim) - mean * mean;
    const float rstd = rsqrtf(var + 1e-5f);

    const int c = tid * 4;
    float4 gv = *(const float4*)(gamma + c);
    float4 bv = *(const float4*)(beta + c);
    bf16_t o[4];
    o[0] = (bf16_t)((xv.x - mean) * rstd * gv.x + bv.x);
    o[1] = (bf16_t)((xv.y - mean) * rstd * gv.y + bv.y);
    o[2] = (bf16_t)((xv.z - mean) * rstd * gv.z + bv.z);
    o[3] = (bf16_t)((xv.w - mean) * rstd * gv.w + bv.w);

    size_t baseA = ((size_t)batch * batchStrideA + rowOffA + r) * kDim + c;
    #pragma unroll
    for (int j = 0; j < 4; ++j) dstA[baseA + j] = o[j];
    if (dstB) {
        size_t baseB = ((size_t)batch * batchStrideB + r) * kDim + c;
        #pragma unroll
        for (int j = 0; j < 4; ++j) dstB[baseB + j] = o[j];
    }
}

// ---------------------------------------------------------------------------
// bf16 GEMM, TN form: C[M,N] = scale * (A[M,K] @ Bt[N,K]^T).
// No LDS: A- and B-fragments are contiguous b128 global loads straight into
// the documented WMMA register layouts; redundant reads hit the 192MB L2.
// Block tile 64x256; 8 waves of 32x64 (2x4 frags), 8 WMMAs per K-step.
// ---------------------------------------------------------------------------
template <int OUTF32>
__global__ __launch_bounds__(256)
void gemm_bf16_tn_kernel(const bf16_t* __restrict__ A,   // M x K row-major
                         const bf16_t* __restrict__ Bt,  // N x K row-major
                         void* __restrict__ Cout, int M, int N, int K, float scale) {
    const int tid  = threadIdx.x;
    const int lane = tid & 31;
    const int wid  = tid >> 5;
    const int waveM = wid >> 2;     // 0..1
    const int waveN = wid & 3;      // 0..3
    const int blockM = blockIdx.y * 64;
    const int blockN = blockIdx.x * 256;

    const int l16 = lane & 15;
    const int hi  = lane >> 4;
    const int b0  = hi ? 8 : 0;

    v8f acc[2][4] = {};

    const bf16_t* aRow[2];
    const bf16_t* bRow[4];
    #pragma unroll
    for (int mi = 0; mi < 2; ++mi)
        aRow[mi] = A + (size_t)(blockM + waveM * 32 + mi * 16 + l16) * K;
    #pragma unroll
    for (int ni = 0; ni < 4; ++ni)
        bRow[ni] = Bt + (size_t)(blockN + waveN * 64 + ni * 16 + l16) * K;

    for (int k0 = 0; k0 < K; k0 += 32) {
        Frag16 af[2], bfr[4];
        #pragma unroll
        for (int mi = 0; mi < 2; ++mi) {
            af[mi].u[0] = *(const uint4*)(aRow[mi] + k0 + b0);
            af[mi].u[1] = *(const uint4*)(aRow[mi] + k0 + b0 + 16);
        }
        const int kb = hi * 8;
        #pragma unroll
        for (int ni = 0; ni < 4; ++ni) {
            bfr[ni].u[0] = *(const uint4*)(bRow[ni] + k0 + kb);
            bfr[ni].u[1] = *(const uint4*)(bRow[ni] + k0 + kb + 16);
        }
        #pragma unroll
        for (int mi = 0; mi < 2; ++mi)
            #pragma unroll
            for (int ni = 0; ni < 4; ++ni)
                acc[mi][ni] = __builtin_amdgcn_wmma_f32_16x16x32_bf16(
                    false, af[mi].v, false, bfr[ni].v, (short)0, acc[mi][ni],
                    false, false);
    }

    #pragma unroll
    for (int mi = 0; mi < 2; ++mi) {
        #pragma unroll
        for (int ni = 0; ni < 4; ++ni) {
            #pragma unroll
            for (int i = 0; i < 8; ++i) {
                const int row = blockM + waveM * 32 + mi * 16 + i + 8 * hi;
                const int col = blockN + waveN * 64 + ni * 16 + l16;
                const float v = acc[mi][ni][i] * scale;
                if (OUTF32)
                    ((float*)Cout)[(size_t)row * N + col] = v;
                else
                    ((bf16_t*)Cout)[(size_t)row * N + col] = (bf16_t)v;
            }
        }
    }
}

// ---------------------------------------------------------------------------
// Flash attention: block = (q-chunk of 128, head, batch); 8 waves x 16 rows.
// 64-key tiles staged through LDS (V transposed); Q pre-scaled by 1/sqrt(d).
// Online softmax with permlane16 row reductions; P restaged via per-wave LDS.
// ---------------------------------------------------------------------------
__global__ __launch_bounds__(256)
void attn_kernel(const bf16_t* __restrict__ Q, const bf16_t* __restrict__ KVbuf,
                 bf16_t* __restrict__ Obuf) {
    using namespace cfg;
    __shared__ __align__(16) bf16_t Ks[64][72];      // [key][d]
    __shared__ __align__(16) bf16_t Vt[64][72];      // [d][key] transposed
    __shared__ __align__(16) bf16_t Ps[8][16][64];   // per-wave probs

    const int tid  = threadIdx.x;
    const int lane = tid & 31;
    const int wid  = tid >> 5;
    const int l16  = lane & 15;
    const int hi   = lane >> 4;
    const int b0   = hi ? 8 : 0;

    const int bz   = blockIdx.z;
    const int head = blockIdx.y;
    const int q0   = blockIdx.x * 128 + wid * 16;

    // Q fragments for this wave's 16 rows (dims 0..31 and 32..63), kept live.
    Frag16 qf[2];
    {
        const bf16_t* qp = Q + ((size_t)(bz * kNl + q0 + l16)) * kDim + head * kDHead;
        #pragma unroll
        for (int f = 0; f < 2; ++f) {
            qf[f].u[0] = *(const uint4*)(qp + f * 32 + b0);
            qf[f].u[1] = *(const uint4*)(qp + f * 32 + b0 + 16);
        }
    }

    float m[8], l[8];
    v8f acc[4] = {};
    #pragma unroll
    for (int i = 0; i < 8; ++i) { m[i] = -1e30f; l[i] = 0.f; }

    const int rk = tid >> 2;         // key row 0..63
    const int ck = (tid & 3) * 16;   // d col 0,16,32,48

    for (int s0 = 0; s0 < kNs; s0 += 64) {
        // Stage K tile (row-major) and V tile (transposed)
        const bf16_t* kp =
            KVbuf + ((size_t)(bz * kNs + s0 + rk)) * (2 * kDim) + head * kDHead + ck;
        *(uint4*)&Ks[rk][ck]     = *(const uint4*)kp;
        *(uint4*)&Ks[rk][ck + 8] = *(const uint4*)(kp + 8);
        Frag16 vu;
        vu.u[0] = *(const uint4*)(kp + kDim);
        vu.u[1] = *(const uint4*)(kp + kDim + 8);
        #pragma unroll
        for (int e = 0; e < 16; ++e) Vt[ck + e][rk] = vu.e[e];
        __syncthreads();

        // Scores: four 16-key subtiles, 2 WMMAs (d=0..31, 32..63) each
        v8f sc[4] = {};
        #pragma unroll
        for (int kt = 0; kt < 4; ++kt) {
            #pragma unroll
            for (int f = 0; f < 2; ++f) {
                Frag16 kf;
                const int kd = f * 32 + hi * 8;
                kf.u[0] = *(const uint4*)&Ks[kt * 16 + l16][kd];
                kf.u[1] = *(const uint4*)&Ks[kt * 16 + l16][kd + 16];
                sc[kt] = __builtin_amdgcn_wmma_f32_16x16x32_bf16(
                    false, qf[f].v, false, kf.v, (short)0, sc[kt], false, false);
            }
        }

        // Online softmax across the 64 new keys; rows live per half-wave.
        #pragma unroll
        for (int i = 0; i < 8; ++i) {
            float sv[4] = { sc[0][i], sc[1][i], sc[2][i], sc[3][i] };
            float mx = fmaxf(fmaxf(sv[0], sv[1]), fmaxf(sv[2], sv[3]));
            mx = red_max16(mx);
            const float mNew  = fmaxf(m[i], mx);
            const float alpha = __expf(m[i] - mNew);
            float p[4], rs = 0.f;
            #pragma unroll
            for (int kt = 0; kt < 4; ++kt) { p[kt] = __expf(sv[kt] - mNew); rs += p[kt]; }
            rs = red_add16(rs);
            l[i] = l[i] * alpha + rs;
            m[i] = mNew;
            #pragma unroll
            for (int t = 0; t < 4; ++t) acc[t][i] *= alpha;
            const int prow = i + 8 * hi;
            #pragma unroll
            for (int kt = 0; kt < 4; ++kt)
                Ps[wid][prow][kt * 16 + l16] = (bf16_t)p[kt];
        }
        // Same-wave LDS RAW: wait for our ds stores before re-reading P.
        asm volatile("s_wait_dscnt 0" ::: "memory");

        Frag16 pf[2];
        #pragma unroll
        for (int f = 0; f < 2; ++f) {
            pf[f].u[0] = *(const uint4*)&Ps[wid][l16][f * 32 + b0];
            pf[f].u[1] = *(const uint4*)&Ps[wid][l16][f * 32 + b0 + 16];
        }

        #pragma unroll
        for (int f = 0; f < 2; ++f) {
            #pragma unroll
            for (int t = 0; t < 4; ++t) {
                Frag16 vf;
                const int kk = f * 32 + hi * 8;
                vf.u[0] = *(const uint4*)&Vt[t * 16 + l16][kk];
                vf.u[1] = *(const uint4*)&Vt[t * 16 + l16][kk + 16];
                acc[t] = __builtin_amdgcn_wmma_f32_16x16x32_bf16(
                    false, pf[f].v, false, vf.v, (short)0, acc[t], false, false);
            }
        }
        __syncthreads();
    }

    #pragma unroll
    for (int t = 0; t < 4; ++t) {
        #pragma unroll
        for (int i = 0; i < 8; ++i) {
            const int row = q0 + i + 8 * hi;
            const int col = head * kDHead + t * 16 + l16;
            const float v = acc[t][i] / l[i];
            Obuf[((size_t)(bz * kNl + row)) * kDim + col] = (bf16_t)v;
        }
    }
}

// ---------------------------------------------------------------------------
// Launch
// ---------------------------------------------------------------------------
extern "C" void kernel_launch(void* const* d_in, const int* in_sizes, int n_in,
                              void* d_out, int out_size, void* d_ws, size_t ws_size,
                              hipStream_t stream) {
    using namespace cfg;
    const float* x    = (const float*)d_in[0];
    const float* lat  = (const float*)d_in[1];
    const float* ln1w = (const float*)d_in[2];
    const float* ln1b = (const float*)d_in[3];
    const float* ln2w = (const float*)d_in[4];
    const float* ln2b = (const float*)d_in[5];
    const float* Wq   = (const float*)d_in[6];
    const float* Wkv  = (const float*)d_in[7];
    const float* Wo   = (const float*)d_in[8];

    char* ws = (char*)d_ws;
    size_t off = 0;
    auto take = [&](size_t bytes) {
        char* p = ws + off;
        off += (bytes + 255) & ~(size_t)255;
        return p;
    };
    bf16_t* xcat = (bf16_t*)take((size_t)kB * kNs * kDim * 2);     // LN(x) ++ LN(lat)
    bf16_t* latb = (bf16_t*)take((size_t)kB * kNl * kDim * 2);     // LN(lat) alone
    bf16_t* wqt  = (bf16_t*)take((size_t)kDim * kDim * 2);         // Wq^T  (N x K)
    bf16_t* wkvt = (bf16_t*)take((size_t)kDim * 2 * kDim * 2);     // Wkv^T (2N x K)
    bf16_t* wot  = (bf16_t*)take((size_t)kDim * kDim * 2);         // Wo^T  (N x K)
    bf16_t* qb   = (bf16_t*)take((size_t)kB * kNl * kDim * 2);     // q * (1/sqrt(d))
    bf16_t* kvb  = (bf16_t*)take((size_t)kB * kNs * 2 * kDim * 2); // [k | v]
    bf16_t* atb  = (bf16_t*)take((size_t)kB * kNl * kDim * 2);     // attention out

    // Weight transpose+convert (K x N f32 -> N x K bf16)
    transpose_f32_bf16_kernel<<<dim3(kDim / 32, kDim / 32), 256, 0, stream>>>(
        Wq, wqt, kDim, kDim);
    transpose_f32_bf16_kernel<<<dim3(kDim / 32, 2 * kDim / 32), 256, 0, stream>>>(
        Wkv, wkvt, kDim, 2 * kDim);
    transpose_f32_bf16_kernel<<<dim3(kDim / 32, kDim / 32), 256, 0, stream>>>(
        Wo, wot, kDim, kDim);

    // LayerNorms (x -> xcat[0:4096), latents -> xcat[4096:5120) and latb)
    ln_to_bf16_kernel<<<kB * kNx, 256, 0, stream>>>(x, ln1w, ln1b,
                                                    xcat, kNs, 0, nullptr, 0, kNx);
    ln_to_bf16_kernel<<<kB * kNl, 256, 0, stream>>>(lat, ln2w, ln2b,
                                                    xcat, kNs, kNx, latb, kNl, kNl);

    // q = LN(lat) @ Wq, both attention scales folded in: (1/d^(1/4))^2 = 0.125
    gemm_bf16_tn_kernel<0><<<dim3(kDim / 256, kB * kNl / 64), 256, 0, stream>>>(
        latb, wqt, qb, kB * kNl, kDim, kDim, 0.125f);
    // kv = [LN(x); LN(lat)] @ Wkv
    gemm_bf16_tn_kernel<0><<<dim3(2 * kDim / 256, kB * kNs / 64), 256, 0, stream>>>(
        xcat, wkvt, kvb, kB * kNs, 2 * kDim, kDim, 1.0f);
    // attention
    attn_kernel<<<dim3(kNl / 128, kHeads, kB), 256, 0, stream>>>(qb, kvb, atb);
    // out = attn @ Wo  (f32 output)
    gemm_bf16_tn_kernel<1><<<dim3(kDim / 256, kB * kNl / 64), 256, 0, stream>>>(
        atb, wot, d_out, kB * kNl, kDim, kDim, 1.0f);
}
